// VarianceAdaptor_34282428957153
// MI455X (gfx1250) — matrix operand order repaired and checked by reading, not compiled, not run on hardware
//
#include <hip/hip_runtime.h>
#include <hip/hip_bf16.h>
#include <stddef.h>

#define BB    32
#define TT    1024
#define HH    384            // input features == F
#define FF    384            // conv output features
#define MM    (BB * TT)      // 32768 rows
#define KC    (3 * FF)       // 1152 im2col K
#define MAXL  2048
#define NBINS 255

typedef __attribute__((ext_vector_type(16))) __bf16 v16bf;
typedef __attribute__((ext_vector_type(8)))  __bf16 v8bf;
typedef __attribute__((ext_vector_type(8)))  float  v8f;

// ---------------- utility kernels ----------------

__global__ void cast_bf16_kernel(const float* __restrict__ src,
                                 __bf16* __restrict__ dst, int n) {
  int i = blockIdx.x * blockDim.x + threadIdx.x;
  if (i < n) dst[i] = (__bf16)src[i];
}

__global__ void zero_bf16_kernel(__bf16* __restrict__ dst, int n) {
  int i = blockIdx.x * blockDim.x + threadIdx.x;
  if (i < n) dst[i] = (__bf16)0.0f;
}

// w: f32 [3][Cin][Cout] -> wt: bf16 [Cout][3*Cin]
__global__ void transpose_w_kernel(const float* __restrict__ w,
                                   __bf16* __restrict__ wt,
                                   int Cin, int Cout) {
  int i = blockIdx.x * blockDim.x + threadIdx.x;
  int n = 3 * Cin * Cout;
  if (i >= n) return;
  int nOut = i % Cout;
  int kc   = i / Cout;           // tap*Cin + c
  wt[(size_t)nOut * (3 * Cin) + kc] = (__bf16)w[i];
}

__device__ __forceinline__ int bin_search_left(const float* __restrict__ bins, float v) {
  int lo = 0, hi = NBINS;
  while (lo < hi) {
    int mid = (lo + hi) >> 1;
    if (bins[mid] < v) lo = mid + 1; else hi = mid;
  }
  return lo;                      // in [0, NBINS] -> valid emb row index
}

// dst_bf16[row] = bf16( x[row] + emb[searchsorted(bins, tgt[row])] )
__global__ __launch_bounds__(128)
void add_emb_cast_kernel(const float* __restrict__ x,
                         const float* __restrict__ tgt,
                         const float* __restrict__ bins,
                         const float* __restrict__ emb,
                         __bf16* __restrict__ dst) {
  int row = blockIdx.x;
  int idx = bin_search_left(bins, tgt[row]);
  const float* xr = x   + (size_t)row * HH;
  const float* er = emb + (size_t)idx * HH;
  __bf16* dr = dst + (size_t)row * HH;
  for (int f = threadIdx.x; f < HH; f += 128)
    dr[f] = (__bf16)(xr[f] + er[f]);
}

// x3 = x + pitch_emb[ip] + energy_emb[ie]  (f32)
__global__ __launch_bounds__(128)
void x3_kernel(const float* __restrict__ x,
               const float* __restrict__ ptgt, const float* __restrict__ pbins,
               const float* __restrict__ pemb,
               const float* __restrict__ etgt, const float* __restrict__ ebins,
               const float* __restrict__ eemb,
               float* __restrict__ dst) {
  int row = blockIdx.x;
  int ip = bin_search_left(pbins, ptgt[row]);
  int ie = bin_search_left(ebins, etgt[row]);
  const float* xr = x    + (size_t)row * HH;
  const float* pr = pemb + (size_t)ip  * HH;
  const float* er = eemb + (size_t)ie  * HH;
  float* dr = dst + (size_t)row * HH;
  for (int f = threadIdx.x; f < HH; f += 128)
    dr[f] = xr[f] + pr[f] + er[f];
}

// ---------------- WMMA conv-as-GEMM ----------------
// A: bf16 [MM][FF] compact activations (taps handled via row shift)
// Wt: bf16 [FF][KC] pre-transposed weights (row n = output channel)
// zrow: bf16 [FF] all-zero row; out-of-range taps redirect here (no cndmask)
// Y: f32 [MM][FF] raw conv output (bias added in epilogue)
//
// Software-pipelined: 36 flattened (tap,k0) steps, double-buffered fragments,
// next step's 10 loads issued before current step's 4 WMMAs so the scheduler
// can overlap load latency with the matrix pipe (partial s_wait_loadcnt).
__global__ __launch_bounds__(256)
void gemm_conv_kernel(const __bf16* __restrict__ A,
                      const __bf16* __restrict__ Wt,
                      const __bf16* __restrict__ zrow,
                      float* __restrict__ Y) {
  const int lane  = threadIdx.x & 31;
  const int wave  = threadIdx.x >> 5;
  const int lr    = lane & 15;     // row within M-tile / col within N-tile
  const int hi    = lane >> 4;     // K sub-chunk selector
  const int mtile = (blockIdx.x * 8 + wave) * 16;
  const int ntile = blockIdx.y * 64;

  const int m     = mtile + lr;          // A row for this lane
  const int t     = m & (TT - 1);        // position within sequence
  const int bbase = m - t;               // batch row base

  // Per-tap A row pointers (invalid taps -> zero row; no per-step masking).
  const __bf16* arows[3];
#pragma unroll
  for (int tap = 0; tap < 3; ++tap) {
    int  tp    = t + tap - 1;
    bool valid = (tp >= 0) && (tp < TT);
    int  tc    = tp < 0 ? 0 : (tp >= TT ? TT - 1 : tp);
    arows[tap] = valid ? (A + (size_t)(bbase + tc) * FF) : zrow;
  }
  // Per-column-tile B row bases.
  const __bf16* wbase[4];
#pragma unroll
  for (int j = 0; j < 4; ++j)
    wbase[j] = Wt + (size_t)(ntile + 16 * j + lr) * KC;

  const int hi8  = hi * 8;
  const int hi16 = hi * 16;

  v8f acc[4] = {v8f{}, v8f{}, v8f{}, v8f{}};
  v16bf aF[2];
  v16bf bF[2][4];

  auto ld = [&](int s, int buf) {
    const int tap = s / 12;
    const int k0  = (s % 12) * 32;
    const __bf16* arow = arows[tap];
    // A fragment: lane<16: K k0..k0+15; lane>=16: K k0+16..k0+31 (contig).
    v8bf alo = *(const v8bf*)(arow + k0 + hi16);
    v8bf ahi = *(const v8bf*)(arow + k0 + hi16 + 8);
#pragma unroll
    for (int i = 0; i < 8; ++i) { aF[buf][i] = alo[i]; aF[buf][i + 8] = ahi[i]; }
#pragma unroll
    for (int j = 0; j < 4; ++j) {
      // B fragment: lane<16: K k0..k0+15; lane>=16: K k0+16..k0+31 (contig).
      const __bf16* wrow = wbase[j] + (size_t)tap * FF + k0 + hi16;
      v8bf b0 = *(const v8bf*)(wrow);
      v8bf b1 = *(const v8bf*)(wrow + 8);
#pragma unroll
      for (int i = 0; i < 8; ++i) { bF[buf][j][i] = b0[i]; bF[buf][j][i + 8] = b1[i]; }
    }
  };

  ld(0, 0);
#pragma unroll
  for (int s = 0; s < 36; ++s) {
    const int cur = s & 1;
    if (s + 1 < 36) ld(s + 1, cur ^ 1);      // prefetch next fragments
#pragma unroll
    for (int j = 0; j < 4; ++j)
      acc[j] = __builtin_amdgcn_wmma_f32_16x16x32_bf16(
          false, aF[cur], false, bF[cur][j], (short)0, acc[j], false, false);
  }

  // D layout: VGPR r -> lanes 0-15 row mtile+r, lanes 16-31 row mtile+8+r
#pragma unroll
  for (int j = 0; j < 4; ++j) {
    int col = ntile + 16 * j + lr;
#pragma unroll
    for (int r = 0; r < 8; ++r) {
      int row = mtile + hi * 8 + r;
      Y[(size_t)row * FF + col] = acc[j][r];
    }
  }
}

// ---------------- fused epilogues ----------------

// Hb = bf16( LN( relu(Y + bias), g, be ) )
__global__ __launch_bounds__(128)
void relu_ln_kernel(const float* __restrict__ Y,
                    const float* __restrict__ bias,
                    const float* __restrict__ g,
                    const float* __restrict__ be,
                    __bf16* __restrict__ H) {
  __shared__ float s1[128], s2[128];
  int row = blockIdx.x;
  int tid = threadIdx.x;
  const float* yr = Y + (size_t)row * FF;
  float v[3], a = 0.f, b2 = 0.f;
#pragma unroll
  for (int j = 0; j < 3; ++j) {
    int f = tid + j * 128;
    float y = yr[f] + bias[f];
    y = y > 0.f ? y : 0.f;
    v[j] = y; a += y; b2 += y * y;
  }
  s1[tid] = a; s2[tid] = b2; __syncthreads();
  for (int s = 64; s > 0; s >>= 1) {
    if (tid < s) { s1[tid] += s1[tid + s]; s2[tid] += s2[tid + s]; }
    __syncthreads();
  }
  float mean = s1[0] * (1.0f / FF);
  float var  = s2[0] * (1.0f / FF) - mean * mean;
  float rs   = rsqrtf(var + 1e-5f);
  __bf16* hr = H + (size_t)row * FF;
#pragma unroll
  for (int j = 0; j < 3; ++j) {
    int f = tid + j * 128;
    hr[f] = (__bf16)((v[j] - mean) * rs * g[f] + be[f]);
  }
}

// out[row] = mask ? 0 : dot( LN(relu(Y+bias), g, be), wo ) + bo
__global__ __launch_bounds__(128)
void final_pred_kernel(const float* __restrict__ Y,
                       const float* __restrict__ bias,
                       const float* __restrict__ g,
                       const float* __restrict__ be,
                       const float* __restrict__ wo,
                       const float* __restrict__ bo,
                       const unsigned char* __restrict__ mask,
                       float* __restrict__ out) {
  __shared__ float s1[128], s2[128];
  int row = blockIdx.x;
  int tid = threadIdx.x;
  const float* yr = Y + (size_t)row * FF;
  float v[3], a = 0.f, b2 = 0.f;
#pragma unroll
  for (int j = 0; j < 3; ++j) {
    int f = tid + j * 128;
    float y = yr[f] + bias[f];
    y = y > 0.f ? y : 0.f;
    v[j] = y; a += y; b2 += y * y;
  }
  s1[tid] = a; s2[tid] = b2; __syncthreads();
  for (int s = 64; s > 0; s >>= 1) {
    if (tid < s) { s1[tid] += s1[tid + s]; s2[tid] += s2[tid + s]; }
    __syncthreads();
  }
  float mean = s1[0] * (1.0f / FF);
  float var  = s2[0] * (1.0f / FF) - mean * mean;
  float rs   = rsqrtf(var + 1e-5f);
  __syncthreads();                       // all reads of s1[0]/s2[0] done

  float acc = 0.f;
#pragma unroll
  for (int j = 0; j < 3; ++j) {
    int f = tid + j * 128;
    float h = (v[j] - mean) * rs * g[f] + be[f];
    acc += h * wo[f];
  }
  s1[tid] = acc; __syncthreads();
  for (int s = 64; s > 0; s >>= 1) {
    if (tid < s) s1[tid] += s1[tid + s];
    __syncthreads();
  }
  if (tid == 0) out[row] = mask[row] ? 0.f : (s1[0] + bo[0]);
}

// ---------------- length regulator ----------------

__global__ __launch_bounds__(1024)
void cumsum_kernel(const int* __restrict__ dur, int* __restrict__ csum,
                   float* __restrict__ mel_out) {
  __shared__ int s[1024];
  int b = blockIdx.x, tid = threadIdx.x;
  s[tid] = dur[b * TT + tid];
  __syncthreads();
  for (int off = 1; off < 1024; off <<= 1) {
    int v = (tid >= off) ? s[tid - off] : 0;
    __syncthreads();
    s[tid] += v;
    __syncthreads();
  }
  csum[b * TT + tid] = s[tid];
  if (tid == TT - 1) mel_out[b] = (float)s[tid];
}

__global__ __launch_bounds__(128)
void gather_kernel(const float* __restrict__ X3,
                   const int* __restrict__ csum,
                   float* __restrict__ out) {
  int blk = blockIdx.x;
  int b   = blk >> 11;                 // / MAXL
  int mfr = blk & (MAXL - 1);
  const int* cs = csum + b * TT;
  int  total = cs[TT - 1];
  bool valid = mfr < total;
  // searchsorted(cs, mfr, side='right'): first idx with cs[idx] > mfr
  int lo = 0, hi = TT;
  while (lo < hi) {
    int mid = (lo + hi) >> 1;
    if (cs[mid] <= mfr) lo = mid + 1; else hi = mid;
  }
  int row = lo > TT - 1 ? TT - 1 : lo;
  const float* src = X3 + ((size_t)b * TT + row) * HH;
  float* dst = out + ((size_t)b * MAXL + mfr) * HH;
  for (int f = threadIdx.x; f < HH; f += 128)
    dst[f] = valid ? src[f] : 0.0f;
}

// ---------------- launcher ----------------

extern "C" void kernel_launch(void* const* d_in, const int* in_sizes, int n_in,
                              void* d_out, int out_size, void* d_ws, size_t ws_size,
                              hipStream_t stream) {
  const float*         x     = (const float*)d_in[0];
  const unsigned char* mask  = (const unsigned char*)d_in[1];
  const int*           dur   = (const int*)d_in[2];
  const float*         pbins = (const float*)d_in[34];
  const float*         ebins = (const float*)d_in[35];
  const float*         pemb  = (const float*)d_in[36];
  const float*         eemb  = (const float*)d_in[37];
  const float*         ptgt  = (const float*)d_in[38];
  const float*         etgt  = (const float*)d_in[39];

  char*  ws  = (char*)d_ws;
  size_t off = 0;
  auto alloc = [&](size_t bytes) -> void* {
    void* p = ws + off;
    off += (bytes + 255) & ~(size_t)255;
    return p;
  };
  __bf16* xb   = (__bf16*)alloc((size_t)MM * HH * 2);
  __bf16* x2b  = (__bf16*)alloc((size_t)MM * HH * 2);
  __bf16* w1t  = (__bf16*)alloc((size_t)FF * KC * 2);
  __bf16* w2t  = (__bf16*)alloc((size_t)FF * KC * 2);
  float*  Y    = (float*)alloc((size_t)MM * FF * 4);
  __bf16* Hb   = (__bf16*)alloc((size_t)MM * FF * 2);
  int*    csum = (int*)alloc((size_t)MM * 4);
  __bf16* zrow = (__bf16*)alloc((size_t)FF * 2);

  float* out_x   = (float*)d_out;
  float* out_pp  = out_x  + (size_t)BB * MAXL * HH;
  float* out_ep  = out_pp + MM;
  float* out_dur = out_ep + MM;
  float* out_len = out_dur + MM;

  {
    int n = MM * HH;
    cast_bf16_kernel<<<(n + 255) / 256, 256, 0, stream>>>(x, xb, n);
    zero_bf16_kernel<<<(FF + 255) / 256, 256, 0, stream>>>(zrow, FF);
  }

  dim3 ggrid(MM / (16 * 8), FF / 64);   // (256, 6)

  auto run_pred = [&](const __bf16* Ain, int base, float* outp) {
    const float* w1  = (const float*)d_in[base + 0];
    const float* b1  = (const float*)d_in[base + 1];
    const float* g1  = (const float*)d_in[base + 2];
    const float* be1 = (const float*)d_in[base + 3];
    const float* w2  = (const float*)d_in[base + 4];
    const float* b2  = (const float*)d_in[base + 5];
    const float* g2  = (const float*)d_in[base + 6];
    const float* be2 = (const float*)d_in[base + 7];
    const float* wo  = (const float*)d_in[base + 8];
    const float* bo  = (const float*)d_in[base + 9];
    int wn = 3 * HH * FF;
    transpose_w_kernel<<<(wn + 255) / 256, 256, 0, stream>>>(w1, w1t, HH, FF);
    transpose_w_kernel<<<(wn + 255) / 256, 256, 0, stream>>>(w2, w2t, FF, FF);
    gemm_conv_kernel<<<ggrid, 256, 0, stream>>>(Ain, w1t, zrow, Y);
    relu_ln_kernel<<<MM, 128, 0, stream>>>(Y, b1, g1, be1, Hb);
    gemm_conv_kernel<<<ggrid, 256, 0, stream>>>(Hb, w2t, zrow, Y);
    final_pred_kernel<<<MM, 128, 0, stream>>>(Y, b2, g2, be2, wo, bo, mask, outp);
  };

  run_pred(xb, 4, out_dur);                                        // duration predictor
  run_pred(xb, 14, out_pp);                                        // pitch predictor
  add_emb_cast_kernel<<<MM, 128, 0, stream>>>(x, ptgt, pbins, pemb, x2b);
  run_pred(x2b, 24, out_ep);                                       // energy predictor
  x3_kernel<<<MM, 128, 0, stream>>>(x, ptgt, pbins, pemb, etgt, ebins, eemb, Y);
  cumsum_kernel<<<BB, 1024, 0, stream>>>(dur, csum, out_len);
  gather_kernel<<<BB * MAXL, 128, 0, stream>>>(Y, csum, out_x);
}
